// TransformerBlock_816043786468
// MI455X (gfx1250) — matrix operand order repaired and checked by reading, not compiled, not run on hardware
//
#include <hip/hip_runtime.h>
#include <hip/hip_bf16.h>
#include <math.h>

// ---------------------------------------------------------------------------
// D=1024 H=16 DH=64 B=2 NSEQ=128 NSEG=16 DFF=4096, L = 2176.
// Canonical row order: rows [0,256) seq (b*128+l); rows [256,4352) struc
// (256 + b*2048 + o).  d_out = [seq 2*128*1024][struc 2*2048*1024].
// All GEMM B matrices are weights, pre-transposed to [N][K] bf16 so every
// LDS access is a 16-byte op.  Tiles staged via global_load_async_to_lds_b128
// (ASYNCcnt) with double buffering; each wave owns a 32x128 output sub-tile
// so every B operand feeds two WMMAs.
// ---------------------------------------------------------------------------

typedef __bf16 bf16_t;
typedef __attribute__((ext_vector_type(16))) __bf16 v16bf;
typedef __attribute__((ext_vector_type(8)))  float  v8f;

union BF16x16 { v16bf v; uint4 q[2]; unsigned short u[16]; };
union F32x8   { v8f   v; float f[8]; };

__device__ __forceinline__ unsigned short f2bf(float f) {
  unsigned int u = __float_as_uint(f);
  unsigned int r = u + 0x7FFFu + ((u >> 16) & 1u);   // round-to-nearest-even
  return (unsigned short)(r >> 16);
}

// ---- CDNA5 async copy to LDS (tracked by ASYNCcnt) ------------------------
__device__ __forceinline__ void async_ld_b128(unsigned short* lds_dst,
                                              const unsigned short* g_src) {
  asm volatile("global_load_async_to_lds_b128 %0, %1, off"
               :: "v"((unsigned)(unsigned long long)lds_dst),
                  "v"((unsigned long long)g_src)
               : "memory");
}
template <int N> __device__ __forceinline__ void wait_async() {
#if __has_builtin(__builtin_amdgcn_s_wait_asynccnt)
  __builtin_amdgcn_s_wait_asynccnt(N);
#else
  asm volatile("s_wait_asynccnt %0" :: "i"(N) : "memory");
#endif
}

// -------------------- weight transpose-cast: W[K][N]f32 -> Wt[N][K]bf16 ----
__global__ __launch_bounds__(256) void wtrans_kernel(const float* __restrict__ W,
                                                     unsigned short* __restrict__ Wt,
                                                     int K, int N) {
  __shared__ unsigned short t[32][33];
  int nb = blockIdx.x * 32, kb = blockIdx.y * 32;
  int tx = threadIdx.x & 31, ty = threadIdx.x >> 5;   // ty: 0..7
  for (int i = ty; i < 32; i += 8)
    t[i][tx] = f2bf(W[(size_t)(kb + i) * N + nb + tx]);
  __syncthreads();
  for (int i = ty; i < 32; i += 8)
    Wt[(size_t)(nb + i) * K + kb + tx] = t[tx][i];
}

// ------------------- row LayerNorm (weight+bias) -> bf16 -------------------
__global__ __launch_bounds__(256) void ln_cast_kernel(const float* __restrict__ X,
                                                      const float* __restrict__ w,
                                                      const float* __restrict__ bias,
                                                      unsigned short* __restrict__ Y) {
  int row = blockIdx.x, tid = threadIdx.x;
  const float* x = X + (size_t)row * 1024;
  float v4[4]; float s = 0.f, ss = 0.f;
  for (int t = 0; t < 4; t++) { float v = x[tid + t * 256]; v4[t] = v; s += v; ss += v * v; }
  for (int m = 1; m < 32; m <<= 1) { s += __shfl_xor(s, m, 32); ss += __shfl_xor(ss, m, 32); }
  __shared__ float a0[8], a1[8];
  if ((tid & 31) == 0) { a0[tid >> 5] = s; a1[tid >> 5] = ss; }
  __syncthreads();
  s = 0.f; ss = 0.f;
  for (int wv = 0; wv < 8; wv++) { s += a0[wv]; ss += a1[wv]; }
  float mu = s * (1.f / 1024.f);
  float var = ss * (1.f / 1024.f) - mu * mu;
  float rs = rsqrtf(var + 1e-5f);
  unsigned short* y = Y + (size_t)row * 1024;
  for (int t = 0; t < 4; t++) {
    int i = tid + t * 256;
    y[i] = f2bf((v4[t] - mu) * rs * w[i] + bias[i]);
  }
}

// --------------------------- bf16 WMMA GEMM --------------------------------
// C[M,N] = A[M,K] @ Bt[N,K]^T; A,Bt bf16 row-major.  EPI: 0 f32 C, 1 gelu->bf16
// block 128 thr (4 waves); tile BM=128 BN=128 BK=32; wave w owns rows
// [w*32, w*32+32) x 128 cols (two 16-row sub-tiles share each B operand).
// Double-buffered async LDS staging (8 x b128 per thread per K-tile).
template <int EPI>
__global__ __launch_bounds__(128) void gemm_bf16_kernel(const unsigned short* __restrict__ A, int lda,
                                                        const unsigned short* __restrict__ Bt,
                                                        void* __restrict__ C, int ldc,
                                                        int M, int N, int K) {
  __shared__ unsigned short As[2][128 * 32];   // [m][k]
  __shared__ unsigned short Bs[2][128 * 32];   // [n][k]
  int tid = threadIdx.x;
  int wave = tid >> 5, lane = tid & 31;
  int half = lane >> 4, l15 = lane & 15;
  int nBase = blockIdx.x * 128;
  int mBase = blockIdx.y * 128;

  F32x8 acc[2][8];
  for (int m = 0; m < 2; m++)
    for (int t = 0; t < 8; t++)
      for (int r = 0; r < 8; r++) acc[m][t].f[r] = 0.f;

  auto stage = [&](int buf, int kt) {
    int k0 = kt * 32;
    const unsigned short* ga = &A[(size_t)(mBase + tid) * lda + k0];
    unsigned short* la = &As[buf][tid * 32];
    async_ld_b128(la,      ga);
    async_ld_b128(la + 8,  ga + 8);
    async_ld_b128(la + 16, ga + 16);
    async_ld_b128(la + 24, ga + 24);
    const unsigned short* gb = &Bt[(size_t)(nBase + tid) * K + k0];
    unsigned short* lb = &Bs[buf][tid * 32];
    async_ld_b128(lb,      gb);
    async_ld_b128(lb + 8,  gb + 8);
    async_ld_b128(lb + 16, gb + 16);
    async_ld_b128(lb + 24, gb + 24);
  };

  int KT = K >> 5;
  stage(0, 0);
  for (int kt = 0; kt < KT; kt++) {
    int buf = kt & 1;
    if (kt + 1 < KT) { stage(buf ^ 1, kt + 1); wait_async<8>(); }
    else             { wait_async<0>(); }
    __syncthreads();
    const unsigned short* Ab = As[buf];
    const unsigned short* Bb = Bs[buf];
    BF16x16 a0, a1;
    { // A operands: rows wave*32+l15 and wave*32+16+l15; K = half*8+{0..7},+16+{0..7}
      const unsigned short* ar0 = &Ab[(wave * 32 + l15) * 32 + half * 8];
      a0.q[0] = *reinterpret_cast<const uint4*>(ar0);
      a0.q[1] = *reinterpret_cast<const uint4*>(ar0 + 16);
      const unsigned short* ar1 = &Ab[(wave * 32 + 16 + l15) * 32 + half * 8];
      a1.q[0] = *reinterpret_cast<const uint4*>(ar1);
      a1.q[1] = *reinterpret_cast<const uint4*>(ar1 + 16);
    }
    for (int t = 0; t < 8; t++) {
      BF16x16 bo;   // col t*16+l15; K = half*8+{0..7},+16+{0..7}
      const unsigned short* br = &Bb[(t * 16 + l15) * 32 + half * 8];
      bo.q[0] = *reinterpret_cast<const uint4*>(br);
      bo.q[1] = *reinterpret_cast<const uint4*>(br + 16);
      acc[0][t].v = __builtin_amdgcn_wmma_f32_16x16x32_bf16(false, a0.v, false, bo.v,
                                                            (short)0, acc[0][t].v, false, false);
      acc[1][t].v = __builtin_amdgcn_wmma_f32_16x16x32_bf16(false, a1.v, false, bo.v,
                                                            (short)0, acc[1][t].v, false, false);
    }
    __syncthreads();
  }
  for (int m = 0; m < 2; m++) {
    for (int t = 0; t < 8; t++) {
      for (int r = 0; r < 8; r++) {
        int rr = mBase + wave * 32 + m * 16 + half * 8 + r;
        int cc = nBase + t * 16 + l15;
        float v = acc[m][t].f[r];
        if (EPI == 0) {
          reinterpret_cast<float*>(C)[(size_t)rr * ldc + cc] = v;
        } else {
          float g = 0.5f * v * (1.0f + erff(v * 0.70710678118654752f));
          reinterpret_cast<unsigned short*>(C)[(size_t)rr * ldc + cc] = f2bf(g);
        }
      }
    }
  }
}

// ------------- qk-LN + RoPE + reorder/concat -> Q,K bf16 [B,H,L,DH], -------
// ------------- V transposed bf16 [B,H,DH,L] (feeds PV B-operand direct) ----
__global__ __launch_bounds__(256) void build_qkv_kernel(const float* __restrict__ qkv, // canonical rows, ld 3072
                                                        const float* __restrict__ qw_seq,
                                                        const float* __restrict__ kw_seq,
                                                        const float* __restrict__ qw_st,
                                                        const float* __restrict__ kw_st,
                                                        unsigned short* __restrict__ Q,
                                                        unsigned short* __restrict__ K,
                                                        unsigned short* __restrict__ Vt) {
  const int L = 2176, H = 16, DH = 64, D = 1024;
  int g = blockIdx.x;
  int b = g / L, l = g % L;
  const float* src; int pos; const float *qw, *kw;
  if (l < 128) {
    src = qkv + (size_t)(b * 128 + l) * 3072;
    pos = l; qw = qw_seq; kw = kw_seq;
  } else {
    int r = l - 128;
    int rs = r >> 7, rr = r & 127;
    int os = (rs == 0) ? 0 : ((rs == 1) ? 15 : rs - 1);   // reordered seg -> original seg
    int o = os * 128 + rr;
    src = qkv + (size_t)(256 + b * 2048 + o) * 3072;
    pos = rr + os;                                        // composed double-RoPE position (T=1)
    qw = qw_st; kw = kw_st;
  }
  int tid = threadIdx.x;
  __shared__ float a0[8], a1[8];
  float stats[2][2];
  for (int part = 0; part < 2; part++) {
    const float* x = src + part * D;
    float s = 0.f, ss = 0.f;
    for (int i = tid; i < D; i += 256) { float v = x[i]; s += v; ss += v * v; }
    for (int m = 1; m < 32; m <<= 1) { s += __shfl_xor(s, m, 32); ss += __shfl_xor(ss, m, 32); }
    if ((tid & 31) == 0) { a0[tid >> 5] = s; a1[tid >> 5] = ss; }
    __syncthreads();
    s = 0.f; ss = 0.f;
    for (int wv = 0; wv < 8; wv++) { s += a0[wv]; ss += a1[wv]; }
    float mu = s * (1.f / 1024.f);
    float var = ss * (1.f / 1024.f) - mu * mu;
    stats[part][0] = mu; stats[part][1] = rsqrtf(var + 1e-5f);
    __syncthreads();
  }
  for (int i = tid; i < D; i += 256) {
    int h = i >> 6, dh = i & 63;
    size_t qkidx = ((size_t)(b * H + h) * L + l) * DH + dh;
    size_t vidx  = ((size_t)(b * H + h) * DH + dh) * L + l;
    Vt[vidx] = f2bf(src[2 * D + i]);
    int j = dh & 31;
    float invf = powf(10000.f, -(float)j * (1.f / 32.f));
    float ang = (float)pos * invf;
    float c = cosf(ang), sn = sinf(ang);
    int pidx = (dh < 32) ? i + 32 : i - 32;
    { // q: LN -> rope -> *1/sqrt(DH)
      float mu = stats[0][0], rs = stats[0][1];
      float y  = (src[i]    - mu) * rs * qw[i];
      float yp = (src[pidx] - mu) * rs * qw[pidx];
      float out = (dh < 32) ? (y * c - yp * sn) : (y * c + yp * sn);
      Q[qkidx] = f2bf(out * 0.125f);
    }
    { // k: LN -> rope
      float mu = stats[1][0], rs = stats[1][1];
      float y  = (src[D + i]    - mu) * rs * kw[i];
      float yp = (src[D + pidx] - mu) * rs * kw[pidx];
      float out = (dh < 32) ? (y * c - yp * sn) : (y * c + yp * sn);
      K[qkidx] = f2bf(out);
    }
  }
}

// -------------------- flash attention, 1 wave / 16 queries -----------------
// grid (L/64, H, B), block 128 (4 waves, each an independent 16-query tile).
// Q,K: [B,H,L,DH]; Vt: [B,H,DH,L]; ctx out bf16 [B*L, 1024] (col = h*64+dh)
__global__ __launch_bounds__(128) void attn_kernel(const unsigned short* __restrict__ Q,
                                                   const unsigned short* __restrict__ Km,
                                                   const unsigned short* __restrict__ Vt,
                                                   const unsigned char* __restrict__ mask,
                                                   unsigned short* __restrict__ ctx) {
  const int L = 2176, H = 16, DH = 64, D = 1024;
  int wave = threadIdx.x >> 5, lane = threadIdx.x & 31;
  int qt = blockIdx.x * 4 + wave;
  int h = blockIdx.y, b = blockIdx.z;
  int half = lane >> 4, l15 = lane & 15;
  int q0 = qt * 16;
  const size_t bhBase = ((size_t)(b * H + h)) * L * DH;
  const size_t vBase  = ((size_t)(b * H + h)) * DH * L;

  __shared__ unsigned short Pl[4][16 * 32];   // per-wave P tile, [q][k]
  unsigned short* P = Pl[wave];

  BF16x16 qa[2];
  {
    const unsigned short* qr = &Q[bhBase + (size_t)(q0 + l15) * DH];
    for (int c = 0; c < 2; c++) {
      qa[c].q[0] = *reinterpret_cast<const uint4*>(qr + c * 32 + half * 8);
      qa[c].q[1] = *reinterpret_cast<const uint4*>(qr + c * 32 + half * 8 + 16);
    }
  }
  // tiled padding mask is periodic mod 128: precompute this lane's 8 bytes
  bool m8[8];
  for (int i = 0; i < 8; i++) m8[i] = mask[b * 128 + ((i * 16 + l15) & 127)] != 0;

  float mrow[8], srow[8];
  F32x8 o[4];
  for (int r = 0; r < 8; r++) { mrow[r] = -3.0e38f; srow[r] = 0.f; }
  for (int t = 0; t < 4; t++) for (int r = 0; r < 8; r++) o[t].f[r] = 0.f;

  int kmax = q0 + 16; if (kmax > L) kmax = L;   // causal bound
  for (int kb = 0; kb < kmax; kb += 32) {
    // S = Q @ K^T for two 16-key groups (B operand = contiguous dims of a key)
    F32x8 s[2];
    for (int g2 = 0; g2 < 2; g2++) {
      for (int r = 0; r < 8; r++) s[g2].f[r] = 0.f;
      int key = kb + g2 * 16 + l15;
      const unsigned short* kr = &Km[bhBase + (size_t)key * DH];
      for (int c = 0; c < 2; c++) {
        BF16x16 kb16;
        kb16.q[0] = *reinterpret_cast<const uint4*>(kr + c * 32 + half * 8);
        kb16.q[1] = *reinterpret_cast<const uint4*>(kr + c * 32 + half * 8 + 16);
        s[g2].v = __builtin_amdgcn_wmma_f32_16x16x32_bf16(false, qa[c].v, false, kb16.v,
                                                          (short)0, s[g2].v, false, false);
      }
    }
    // causal + padding mask
    for (int g2 = 0; g2 < 2; g2++) {
      int key = kb + g2 * 16 + l15;
      bool kok = m8[((kb >> 4) + g2) & 7];
      for (int r = 0; r < 8; r++) {
        int qr_ = q0 + half * 8 + r;
        if (!kok || key > qr_) s[g2].f[r] = -3.0e38f;
      }
    }
    // online softmax update (row stats shared across the 16 lanes of a half)
    for (int r = 0; r < 8; r++) {
      float mx = fmaxf(s[0].f[r], s[1].f[r]);
      for (int x = 1; x < 16; x <<= 1) mx = fmaxf(mx, __shfl_xor(mx, x, 32));
      float mnew = fmaxf(mrow[r], mx);
      float alpha = expf(mrow[r] - mnew);
      mrow[r] = mnew;
      float p0 = expf(s[0].f[r] - mnew);
      float p1 = expf(s[1].f[r] - mnew);
      float psum = p0 + p1;
      for (int x = 1; x < 16; x <<= 1) psum += __shfl_xor(psum, x, 32);
      srow[r] = srow[r] * alpha + psum;
      for (int t = 0; t < 4; t++) o[t].f[r] *= alpha;
      P[(half * 8 + r) * 32 + l15]      = f2bf(p0);
      P[(half * 8 + r) * 32 + 16 + l15] = f2bf(p1);
    }
    // O += P @ V   (V^T rows are contiguous in keys: direct global B-operand)
    BF16x16 pa;
    {
      const unsigned short* pr = &P[l15 * 32 + half * 8];
      pa.q[0] = *reinterpret_cast<const uint4*>(pr);
      pa.q[1] = *reinterpret_cast<const uint4*>(pr + 16);
    }
    for (int t = 0; t < 4; t++) {
      BF16x16 vb;
      const unsigned short* vr = &Vt[vBase + (size_t)(t * 16 + l15) * L + kb + half * 8];
      vb.q[0] = *reinterpret_cast<const uint4*>(vr);
      vb.q[1] = *reinterpret_cast<const uint4*>(vr + 16);
      o[t].v = __builtin_amdgcn_wmma_f32_16x16x32_bf16(false, pa.v, false, vb.v,
                                                       (short)0, o[t].v, false, false);
    }
  }
  for (int t = 0; t < 4; t++) {
    for (int r = 0; r < 8; r++) {
      float denom = srow[r];
      float val = (denom > 0.f) ? o[t].f[r] / denom : 0.f;
      int qr_ = q0 + half * 8 + r;
      ctx[((size_t)(b * L + qr_)) * D + h * DH + t * 16 + l15] = f2bf(val);
    }
  }
}

// ---------------- residual + inverse traj-sample reorder -> d_out ----------
__global__ __launch_bounds__(256) void resid_reorder_kernel(const float* __restrict__ xs,
                                                            const float* __restrict__ xt,
                                                            const float* __restrict__ op,
                                                            float* __restrict__ out) {
  int i = blockIdx.x * 256 + threadIdx.x;      // < 4456448
  if (i < 262144) {
    out[i] = xs[i] + op[i];
  } else {
    int gs = i - 262144;
    int row = gs >> 10, d = gs & 1023;         // row = b*2048 + f
    int f = row & 2047;
    int fs = f >> 7;
    int rs = (fs == 0) ? 0 : ((fs == 15) ? 1 : fs + 1);  // final seg -> reordered seg
    int srcrow = 256 + (row - f) + rs * 128 + (f & 127);
    out[i] = xt[gs] + op[(size_t)srcrow * 1024 + d];
  }
}

__global__ __launch_bounds__(256) void final_add_kernel(float* __restrict__ out,
                                                        const float* __restrict__ y, int n) {
  int i = blockIdx.x * 256 + threadIdx.x;
  if (i < n) out[i] += y[i];
}

// ---------------------------------------------------------------------------
extern "C" void kernel_launch(void* const* d_in, const int* in_sizes, int n_in,
                              void* d_out, int out_size, void* d_ws, size_t ws_size,
                              hipStream_t stream) {
  const float* x_seq        = (const float*)d_in[0];
  const float* x_struc      = (const float*)d_in[1];
  const unsigned char* mask = (const unsigned char*)d_in[2];
  const float* ln_qkv_seq_w = (const float*)d_in[3];
  const float* ln_qkv_seq_b = (const float*)d_in[4];
  const float* W_qkv_seq    = (const float*)d_in[5];
  const float* ln_qkv_st_w  = (const float*)d_in[6];
  const float* ln_qkv_st_b  = (const float*)d_in[7];
  const float* W_qkv_st     = (const float*)d_in[8];
  const float* q_ln_seq_w   = (const float*)d_in[9];
  const float* k_ln_seq_w   = (const float*)d_in[10];
  const float* q_ln_st_w    = (const float*)d_in[11];
  const float* k_ln_st_w    = (const float*)d_in[12];
  const float* W_out_seq    = (const float*)d_in[13];
  const float* W_out_st     = (const float*)d_in[14];
  const float* ffn_s_ln_w   = (const float*)d_in[15];
  const float* ffn_s_ln_b   = (const float*)d_in[16];
  const float* ffn_s_W1     = (const float*)d_in[17];
  const float* ffn_s_W2     = (const float*)d_in[18];
  const float* ffn_t_ln_w   = (const float*)d_in[19];
  const float* ffn_t_ln_b   = (const float*)d_in[20];
  const float* ffn_t_W1     = (const float*)d_in[21];
  const float* ffn_t_W2     = (const float*)d_in[22];
  float* out = (float*)d_out;

  char* base = (char*)d_ws;
  size_t off = 0;
  auto take = [&](size_t bytes) -> char* {
    char* p = base + off; off += (bytes + 255) & ~(size_t)255; return p;
  };
  // transposed bf16 weights [N][K]
  unsigned short* Wq_s = (unsigned short*)take((size_t)1024 * 3072 * 2);
  unsigned short* Wq_t = (unsigned short*)take((size_t)1024 * 3072 * 2);
  unsigned short* Wo_s = (unsigned short*)take((size_t)1024 * 1024 * 2);
  unsigned short* Wo_t = (unsigned short*)take((size_t)1024 * 1024 * 2);
  unsigned short* W1_s = (unsigned short*)take((size_t)1024 * 4096 * 2);
  unsigned short* W2_s = (unsigned short*)take((size_t)4096 * 1024 * 2);
  unsigned short* W1_t = (unsigned short*)take((size_t)1024 * 4096 * 2);
  unsigned short* W2_t = (unsigned short*)take((size_t)4096 * 1024 * 2);
  unsigned short* lnA  = (unsigned short*)take((size_t)4352 * 1024 * 2);
  char* big            = take((size_t)4352 * 3072 * 4);       // qkv f32; reused as FFN hidden (bf16)
  float* qkvf          = (float*)big;
  unsigned short* hbuf = (unsigned short*)big;
  unsigned short* Qb   = (unsigned short*)take((size_t)4456448 * 2);
  unsigned short* Kb   = (unsigned short*)take((size_t)4456448 * 2);
  unsigned short* Vb   = (unsigned short*)take((size_t)4456448 * 2);  // transposed [B,H,DH,L]
  unsigned short* ctx  = (unsigned short*)take((size_t)4456448 * 2);
  char* big2           = take((size_t)4456448 * 4);           // out-proj f32; reused as FFN y f32
  float* oproj = (float*)big2;
  float* ybuf  = (float*)big2;
  (void)ws_size; (void)in_sizes; (void)n_in; (void)out_size;

  dim3 blk256(256), blk128(128);
  auto WT = [&](const float* s, unsigned short* d, int K, int N) {
    wtrans_kernel<<<dim3(N / 32, K / 32), blk256, 0, stream>>>(s, d, K, N);
  };
  // 1. weights -> transposed bf16
  WT(W_qkv_seq, Wq_s, 1024, 3072);  WT(W_qkv_st, Wq_t, 1024, 3072);
  WT(W_out_seq, Wo_s, 1024, 1024);  WT(W_out_st, Wo_t, 1024, 1024);
  WT(ffn_s_W1,  W1_s, 1024, 4096);  WT(ffn_s_W2, W2_s, 4096, 1024);
  WT(ffn_t_W1,  W1_t, 1024, 4096);  WT(ffn_t_W2, W2_t, 4096, 1024);
  // 2. LN1 -> bf16 (canonical rows)
  ln_cast_kernel<<<dim3(256),  blk256, 0, stream>>>(x_seq,   ln_qkv_seq_w, ln_qkv_seq_b, lnA);
  ln_cast_kernel<<<dim3(4096), blk256, 0, stream>>>(x_struc, ln_qkv_st_w,  ln_qkv_st_b,  lnA + (size_t)256 * 1024);
  // 3. QKV GEMMs
  gemm_bf16_kernel<0><<<dim3(24, 2),  blk128, 0, stream>>>(lnA,                      1024, Wq_s, qkvf,                       3072, 256,  3072, 1024);
  gemm_bf16_kernel<0><<<dim3(24, 32), blk128, 0, stream>>>(lnA + (size_t)256 * 1024, 1024, Wq_t, qkvf + (size_t)256 * 3072,  3072, 4096, 3072, 1024);
  // 4. qk-LN + RoPE + reorder/concat (V emitted transposed)
  build_qkv_kernel<<<dim3(4352), blk256, 0, stream>>>(qkvf, q_ln_seq_w, k_ln_seq_w, q_ln_st_w, k_ln_st_w, Qb, Kb, Vb);
  // 5. attention
  attn_kernel<<<dim3(34, 16, 2), blk128, 0, stream>>>(Qb, Kb, Vb, mask, ctx);
  // 6. output projections (per batch, per stream; C rows in canonical order)
  gemm_bf16_kernel<0><<<dim3(8, 1),  blk128, 0, stream>>>(ctx,                               1024, Wo_s, oproj,                               1024, 128,  1024, 1024);
  gemm_bf16_kernel<0><<<dim3(8, 1),  blk128, 0, stream>>>(ctx + (size_t)2176 * 1024,         1024, Wo_s, oproj + (size_t)128 * 1024,          1024, 128,  1024, 1024);
  gemm_bf16_kernel<0><<<dim3(8, 16), blk128, 0, stream>>>(ctx + (size_t)128 * 1024,          1024, Wo_t, oproj + (size_t)256 * 1024,          1024, 2048, 1024, 1024);
  gemm_bf16_kernel<0><<<dim3(8, 16), blk128, 0, stream>>>(ctx + (size_t)(2176 + 128) * 1024, 1024, Wo_t, oproj + (size_t)(256 + 2048) * 1024, 1024, 2048, 1024, 1024);
  // 7. residual + inverse reorder -> d_out (= x1)
  resid_reorder_kernel<<<dim3(17408), blk256, 0, stream>>>(x_seq, x_struc, oproj, out);
  // 8. FFN LN -> bf16
  ln_cast_kernel<<<dim3(256),  blk256, 0, stream>>>(out,          ffn_s_ln_w, ffn_s_ln_b, lnA);
  ln_cast_kernel<<<dim3(4096), blk256, 0, stream>>>(out + 262144, ffn_t_ln_w, ffn_t_ln_b, lnA + (size_t)256 * 1024);
  // 9. FFN1 + exact-erf GELU -> bf16 hidden
  gemm_bf16_kernel<1><<<dim3(32, 2),  blk128, 0, stream>>>(lnA,                      1024, W1_s, hbuf,                       4096, 256,  4096, 1024);
  gemm_bf16_kernel<1><<<dim3(32, 32), blk128, 0, stream>>>(lnA + (size_t)256 * 1024, 1024, W1_t, hbuf + (size_t)256 * 4096,  4096, 4096, 4096, 1024);
  // 10. FFN2 -> f32
  gemm_bf16_kernel<0><<<dim3(8, 2),  blk128, 0, stream>>>(hbuf,                      4096, W2_s, ybuf,                       1024, 256,  1024, 4096);
  gemm_bf16_kernel<0><<<dim3(8, 32), blk128, 0, stream>>>(hbuf + (size_t)256 * 4096, 4096, W2_t, ybuf + (size_t)256 * 1024,  1024, 4096, 1024, 4096);
  // 11. final residual
  final_add_kernel<<<dim3(17408), blk256, 0, stream>>>(out, ybuf, 4456448);
}
